// DAWN_71167608094944
// MI455X (gfx1250) — matrix operand order repaired and checked by reading, not compiled, not run on hardware
//
#include <hip/hip_runtime.h>
#include <hip/hip_bf16.h>

// ---------------------------------------------------------------------------
// Types for CDNA5 WMMA (wave32)
// ---------------------------------------------------------------------------
typedef __attribute__((ext_vector_type(16))) __bf16    v16bf;
typedef __attribute__((ext_vector_type(8)))  float     v8f;
typedef __attribute__((ext_vector_type(4)))  unsigned int u32x4;

union Frag {
    v16bf v;
    u32x4 q[2];
    unsigned short s[16];
};

__device__ __forceinline__ unsigned short f2b(float f) {
    unsigned int u = __float_as_uint(f);
    unsigned int r = (u + 0x7FFFu + ((u >> 16) & 1u)) >> 16;
    return (unsigned short)r;
}
__device__ __forceinline__ float b2f(unsigned short s) {
    return __uint_as_float(((unsigned int)s) << 16);
}

__device__ __forceinline__ v8f wmma_bf16(const Frag& a, const Frag& b, v8f c) {
    return __builtin_amdgcn_wmma_f32_16x16x32_bf16(
        false, a.v, false, b.v, (short)0, c, false, false);
}

// Async global->LDS 16-byte copy (gfx1250 GLOBAL_LOAD_ASYNC_TO_LDS_B128,
// tracked on ASYNCcnt). lds_off = low 32 bits of the generic LDS pointer
// (flat->LDS mapping truncates to addr[31:0] per ISA aperture rules).
__device__ __forceinline__ void async_copy_b128(void* lds_dst, const void* gsrc) {
    unsigned lds_off = (unsigned)(unsigned long long)lds_dst;
    asm volatile("global_load_async_to_lds_b128 %0, %1, off"
                 :: "v"(lds_off), "v"(gsrc) : "memory");
}
__device__ __forceinline__ void wait_async0() {
    asm volatile("s_wait_asynccnt 0" ::: "memory");
}

// ---------------------------------------------------------------------------
// Tiled bf16 GEMM:  C[M,N] = epilogue(A[M,K] @ B[K,N])
// Block = 256 threads = 8 waves tiled 4x2 over a BM x BN tile.
// Each wave: (BM/64)x(BN/32) 16x16 fp32 accumulators, 16x16x32 bf16 WMMA.
// A tile staged via async global->LDS (requires M % BM == 0, true at all
// call sites). B tile staged via VGPRs with on-the-fly transpose.
// act: 0=none, 1=gelu(exact), 2=relu.  out = act(acc+bias) + beta*Res
// Batched via blockIdx.z with element strides.
// ---------------------------------------------------------------------------
template<int BM, int BN>
__global__ __launch_bounds__(256) void gemm_bf16_t(
    const unsigned short* __restrict__ A, const unsigned short* __restrict__ B,
    const float* __restrict__ bias, float* __restrict__ Cf,
    unsigned short* __restrict__ Cb, const float* __restrict__ Res,
    float beta, int act, int M, int N, int K,
    long strideA, long strideB, long strideC)
{
    constexpr int SM = BM / 64;   // 16x16 subtile rows per wave
    constexpr int SN = BN / 32;   // 16x16 subtile cols per wave

    int bz = blockIdx.z;
    A += (long)bz * strideA;
    B += (long)bz * strideB;
    long cbase = (long)bz * strideC;

    int m0 = blockIdx.y * BM;
    int n0 = blockIdx.x * BN;

    __shared__ __align__(16) unsigned short Als[BM * 32];
    __shared__ __align__(16) unsigned short Bls[BN * 32];   // transposed: [n][k]

    int tid  = threadIdx.x;
    int lane = tid & 31;
    int wave = tid >> 5;
    int half = lane >> 4;
    int l16  = lane & 15;
    int wrow = (wave >> 1) * (BM / 4);   // wave row base in tile
    int wcol = (wave & 1) * (BN / 2);    // wave col base in tile

    v8f zero = {};
    v8f acc[SM][SN];
#pragma unroll
    for (int i = 0; i < SM; ++i)
#pragma unroll
        for (int j = 0; j < SN; ++j) acc[i][j] = zero;

    for (int kt = 0; kt < K; kt += 32) {
        // ---- stage A tile [BM x 32] via async global->LDS copies ----
#pragma unroll
        for (int e = 0; e < BM * 32; e += 2048) {
            int idx = e + tid * 8;
            int row = idx >> 5, seg = idx & 31;
            async_copy_b128(&Als[idx], A + (long)(m0 + row) * K + kt + seg);
            if (kt + 32 < K)
                __builtin_prefetch((const void*)(A + (long)(m0 + row) * K + kt + 32 + seg), 0, 3);
        }
        // ---- stage B tile [32 x BN] transposed into Bls[n][k] ----
#pragma unroll
        for (int e = 0; e < BN * 32; e += 2048) {
            int idx = e + tid * 8;
            int kr = idx / BN, col = idx % BN;
            u32x4 bv = {};
            if (n0 + col + 8 <= N)
                bv = *reinterpret_cast<const u32x4*>(B + (long)(kt + kr) * N + n0 + col);
            if (kt + 32 < K)
                __builtin_prefetch((const void*)(B + (long)(kt + 32 + kr) * N + n0 + col), 0, 3);
            union { u32x4 q; unsigned short s[8]; } u;
            u.q = bv;
#pragma unroll
            for (int j = 0; j < 8; ++j)
                Bls[(col + j) * 32 + kr] = u.s[j];
        }
        wait_async0();
        __syncthreads();

        Frag fa[SM], fb[SN];
#pragma unroll
        for (int i = 0; i < SM; ++i) {
            int r = wrow + i * 16 + l16;
            // A-frag: lane<16 -> K 0..15; lane>=16 -> K 16..31 (contiguous)
            fa[i].q[0] = *reinterpret_cast<const u32x4*>(&Als[r * 32 + half * 16]);
            fa[i].q[1] = *reinterpret_cast<const u32x4*>(&Als[r * 32 + half * 16 + 8]);
        }
#pragma unroll
        for (int j = 0; j < SN; ++j) {
            int c = wcol + j * 16 + l16;
            // B-frag: lane<16 -> K 0..15; lane>=16 -> K 16..31 (contiguous)
            fb[j].q[0] = *reinterpret_cast<const u32x4*>(&Bls[c * 32 + half * 16]);
            fb[j].q[1] = *reinterpret_cast<const u32x4*>(&Bls[c * 32 + half * 16 + 8]);
        }
#pragma unroll
        for (int i = 0; i < SM; ++i)
#pragma unroll
            for (int j = 0; j < SN; ++j)
                acc[i][j] = wmma_bf16(fa[i], fb[j], acc[i][j]);
        __syncthreads();
    }

    // ---- epilogue: C element (m,n): VGPR v -> m = v + 8*half, lane l16 -> n
#pragma unroll
    for (int i = 0; i < SM; ++i) {
#pragma unroll
        for (int v = 0; v < 8; ++v) {
            int m = m0 + wrow + i * 16 + v + 8 * half;
            if (m >= M) continue;
#pragma unroll
            for (int j = 0; j < SN; ++j) {
                int c = n0 + wcol + j * 16 + l16;
                if (c >= N) continue;
                float xv = acc[i][j][v];
                if (bias) xv += bias[c];
                if (act == 1) xv = 0.5f * xv * (1.0f + erff(xv * 0.70710678f));
                else if (act == 2) xv = fmaxf(xv, 0.0f);
                long idx = cbase + (long)m * N + c;
                if (Res) xv += beta * Res[idx];
                if (Cf) Cf[idx] = xv;
                if (Cb) Cb[idx] = f2b(xv);
            }
        }
    }
}

// ---------------------------------------------------------------------------
// Flash attention: 4 waves/block, 16 queries per wave, 32-key tiles.
// Q,K,V bf16 [B,S,D] with head slice at h*64. Writes ctx bf16.
// K tile staged via async global->LDS; V tile transposed via VGPRs.
// ---------------------------------------------------------------------------
__global__ __launch_bounds__(128) void attn_kernel(
    const unsigned short* __restrict__ Qg, const unsigned short* __restrict__ Kg,
    const unsigned short* __restrict__ Vg, unsigned short* __restrict__ Ob,
    int S, int D, int H)
{
    int bh = blockIdx.y;
    int b = bh / H, h = bh % H;
    const int DH = 64;
    int q0 = blockIdx.x * 64;

    int tid  = threadIdx.x;
    int lane = tid & 31;
    int wave = tid >> 5;
    int half = lane >> 4;
    int l16  = lane & 15;

    long base = ((long)b * S) * D + (long)h * DH;

    __shared__ __align__(16) unsigned short Kls[32 * 64];       // [key][d]
    __shared__ __align__(16) unsigned short Vls[64 * 32];       // [d][key]
    __shared__ __align__(16) float          Scl[4][16 * 32];
    __shared__ __align__(16) unsigned short Pls[4][16 * 32];
    __shared__ float rowm[4][16], rowl[4][16], rowsc[4][16];

    // per-wave Q fragments (K-dim 0..31 and 32..63)
    int qrow = q0 + wave * 16 + l16;
    const unsigned short* qp = Qg + base + (long)qrow * D;
    Frag qa0, qa1;
    qa0.q[0] = *reinterpret_cast<const u32x4*>(qp + half * 16);
    qa0.q[1] = *reinterpret_cast<const u32x4*>(qp + half * 16 + 8);
    qa1.q[0] = *reinterpret_cast<const u32x4*>(qp + 32 + half * 16);
    qa1.q[1] = *reinterpret_cast<const u32x4*>(qp + 32 + half * 16 + 8);

    v8f oc0 = {}, oc1 = {}, oc2 = {}, oc3 = {};
    if (lane < 16) { rowm[wave][lane] = -1e30f; rowl[wave][lane] = 0.0f; }

    int kr   = tid >> 2;            // 0..31
    int dseg = (tid & 3) * 16;      // 0,16,32,48

    for (int kt = 0; kt < S; kt += 32) {
        // stage K row-major via async copies, V transposed via VGPRs
        const unsigned short* kp = Kg + base + (long)(kt + kr) * D + dseg;
        async_copy_b128(&Kls[kr * 64 + dseg], kp);
        async_copy_b128(&Kls[kr * 64 + dseg + 8], kp + 8);
        const unsigned short* vp = Vg + base + (long)(kt + kr) * D + dseg;
        union { u32x4 q; unsigned short s[8]; } u0, u1;
        u0.q = *reinterpret_cast<const u32x4*>(vp);
        u1.q = *reinterpret_cast<const u32x4*>(vp + 8);
#pragma unroll
        for (int j = 0; j < 8; ++j) {
            Vls[(dseg + j) * 32 + kr]     = u0.s[j];
            Vls[(dseg + 8 + j) * 32 + kr] = u1.s[j];
        }
        if (kt + 32 < S) {
            __builtin_prefetch((const void*)(Kg + base + (long)(kt + 32 + kr) * D + dseg), 0, 3);
            __builtin_prefetch((const void*)(Vg + base + (long)(kt + 32 + kr) * D + dseg), 0, 3);
        }
        wait_async0();
        __syncthreads();

        // S_t[16x32] = Q[16x64] @ K_t^T   (B-frag reads contiguous d from Kls)
        v8f s0 = {}, s1 = {};
        Frag fb;
        fb.q[0] = *reinterpret_cast<const u32x4*>(&Kls[l16 * 64 + half * 16]);
        fb.q[1] = *reinterpret_cast<const u32x4*>(&Kls[l16 * 64 + half * 16 + 8]);
        s0 = wmma_bf16(qa0, fb, s0);
        fb.q[0] = *reinterpret_cast<const u32x4*>(&Kls[l16 * 64 + 32 + half * 16]);
        fb.q[1] = *reinterpret_cast<const u32x4*>(&Kls[l16 * 64 + 32 + half * 16 + 8]);
        s0 = wmma_bf16(qa1, fb, s0);
        fb.q[0] = *reinterpret_cast<const u32x4*>(&Kls[(16 + l16) * 64 + half * 16]);
        fb.q[1] = *reinterpret_cast<const u32x4*>(&Kls[(16 + l16) * 64 + half * 16 + 8]);
        s1 = wmma_bf16(qa0, fb, s1);
        fb.q[0] = *reinterpret_cast<const u32x4*>(&Kls[(16 + l16) * 64 + 32 + half * 16]);
        fb.q[1] = *reinterpret_cast<const u32x4*>(&Kls[(16 + l16) * 64 + 32 + half * 16 + 8]);
        s1 = wmma_bf16(qa1, fb, s1);

        // dump scores to LDS (same-wave)
#pragma unroll
        for (int v = 0; v < 8; ++v) {
            Scl[wave][(v + 8 * half) * 32 + l16]      = s0[v];
            Scl[wave][(v + 8 * half) * 32 + 16 + l16] = s1[v];
        }

        // online softmax row pass (one row per lane, lanes 0..15)
        if (lane < 16) {
            int r = lane;
            float om = rowm[wave][r], ol = rowl[wave][r];
            float srow[32];
            float tm = -1e30f;
#pragma unroll
            for (int j = 0; j < 32; ++j) {
                srow[j] = Scl[wave][r * 32 + j] * 0.125f;  // 1/sqrt(64)
                tm = fmaxf(tm, srow[j]);
            }
            float nm = fmaxf(om, tm);
            float scale = __expf(om - nm);
            float ts = 0.0f;
#pragma unroll
            for (int j = 0; j < 32; ++j) {
                float p = __expf(srow[j] - nm);
                ts += p;
                Pls[wave][r * 32 + j] = f2b(p);
            }
            rowm[wave][r]  = nm;
            rowl[wave][r]  = ol * scale + ts;
            rowsc[wave][r] = scale;
        }

        // rescale O accumulators
#pragma unroll
        for (int v = 0; v < 8; ++v) {
            float f = rowsc[wave][v + 8 * half];
            oc0[v] *= f; oc1[v] *= f; oc2[v] *= f; oc3[v] *= f;
        }

        // O += P[16x32] @ V[32x64]
        Frag pa, fv;
        pa.q[0] = *reinterpret_cast<const u32x4*>(&Pls[wave][l16 * 32 + half * 16]);
        pa.q[1] = *reinterpret_cast<const u32x4*>(&Pls[wave][l16 * 32 + half * 16 + 8]);
        fv.q[0] = *reinterpret_cast<const u32x4*>(&Vls[(0 + l16) * 32 + half * 16]);
        fv.q[1] = *reinterpret_cast<const u32x4*>(&Vls[(0 + l16) * 32 + half * 16 + 8]);
        oc0 = wmma_bf16(pa, fv, oc0);
        fv.q[0] = *reinterpret_cast<const u32x4*>(&Vls[(16 + l16) * 32 + half * 16]);
        fv.q[1] = *reinterpret_cast<const u32x4*>(&Vls[(16 + l16) * 32 + half * 16 + 8]);
        oc1 = wmma_bf16(pa, fv, oc1);
        fv.q[0] = *reinterpret_cast<const u32x4*>(&Vls[(32 + l16) * 32 + half * 16]);
        fv.q[1] = *reinterpret_cast<const u32x4*>(&Vls[(32 + l16) * 32 + half * 16 + 8]);
        oc2 = wmma_bf16(pa, fv, oc2);
        fv.q[0] = *reinterpret_cast<const u32x4*>(&Vls[(48 + l16) * 32 + half * 16]);
        fv.q[1] = *reinterpret_cast<const u32x4*>(&Vls[(48 + l16) * 32 + half * 16 + 8]);
        oc3 = wmma_bf16(pa, fv, oc3);

        __syncthreads();
    }

    // finalize: divide by row sum, write ctx bf16
#pragma unroll
    for (int v = 0; v < 8; ++v) {
        float inv = 1.0f / rowl[wave][v + 8 * half];
        int row = q0 + wave * 16 + v + 8 * half;
        long ob = base + (long)row * D;
        Ob[ob + 0  + l16] = f2b(oc0[v] * inv);
        Ob[ob + 16 + l16] = f2b(oc1[v] * inv);
        Ob[ob + 32 + l16] = f2b(oc2[v] * inv);
        Ob[ob + 48 + l16] = f2b(oc3[v] * inv);
    }
}

// ---------------------------------------------------------------------------
// LayerNorm: fp32 in -> bf16 out, one row (D elems) per block
// ---------------------------------------------------------------------------
__global__ __launch_bounds__(256) void ln_kernel(
    const float* __restrict__ x, const float* __restrict__ g,
    const float* __restrict__ be, unsigned short* __restrict__ out, int Dd)
{
    int row = blockIdx.x, tid = threadIdx.x;
    const float* xp = x + (long)row * Dd;
    __shared__ float red[256];
    __shared__ float mean_s, rstd_s;
    float s = 0.0f;
    for (int i = tid; i < Dd; i += 256) s += xp[i];
    red[tid] = s; __syncthreads();
    for (int st = 128; st > 0; st >>= 1) { if (tid < st) red[tid] += red[tid + st]; __syncthreads(); }
    if (tid == 0) mean_s = red[0] / Dd;
    __syncthreads();
    float mean = mean_s, vs = 0.0f;
    for (int i = tid; i < Dd; i += 256) { float d = xp[i] - mean; vs += d * d; }
    red[tid] = vs; __syncthreads();
    for (int st = 128; st > 0; st >>= 1) { if (tid < st) red[tid] += red[tid + st]; __syncthreads(); }
    if (tid == 0) rstd_s = rsqrtf(red[0] / Dd + 1e-5f);
    __syncthreads();
    float rstd = rstd_s;
    unsigned short* op = out + (long)row * Dd;
    for (int i = tid; i < Dd; i += 256) op[i] = f2b((xp[i] - mean) * rstd * g[i] + be[i]);
}

// ---------------------------------------------------------------------------
// Router gate: w = softmax([n, ctx] @ Wp + bp), one token per block
// ---------------------------------------------------------------------------
__global__ __launch_bounds__(256) void wp_kernel(
    const unsigned short* __restrict__ n, const unsigned short* __restrict__ ctx,
    const float* __restrict__ Wp, const float* __restrict__ bp,
    float* __restrict__ w, int Dd)
{
    int tok = blockIdx.x, tid = threadIdx.x;
    const unsigned short* np = n + (long)tok * Dd;
    const unsigned short* cp = ctx + (long)tok * Dd;
    __shared__ float r0[256], r1[256];
    float a0 = 0.0f, a1 = 0.0f;
    for (int i = tid; i < Dd; i += 256) {
        float nv = b2f(np[i]), cv = b2f(cp[i]);
        a0 += nv * Wp[(long)i * 2 + 0] + cv * Wp[(long)(Dd + i) * 2 + 0];
        a1 += nv * Wp[(long)i * 2 + 1] + cv * Wp[(long)(Dd + i) * 2 + 1];
    }
    r0[tid] = a0; r1[tid] = a1; __syncthreads();
    for (int st = 128; st > 0; st >>= 1) {
        if (tid < st) { r0[tid] += r0[tid + st]; r1[tid] += r1[tid + st]; }
        __syncthreads();
    }
    if (tid == 0) {
        float z0 = r0[0] + bp[0], z1 = r1[0] + bp[1];
        float mx = fmaxf(z0, z1);
        float e0 = __expf(z0 - mx), e1 = __expf(z1 - mx);
        float inv = 1.0f / (e0 + e1);
        w[(long)tok * 2 + 0] = e0 * inv;
        w[(long)tok * 2 + 1] = e1 * inv;
    }
}

// t1c = bf16(w0*t1_tok + w1*t1_ctx)
__global__ void combine_t1_kernel(const float* __restrict__ t1t,
    const float* __restrict__ t1c, const float* __restrict__ w,
    unsigned short* __restrict__ out, int NR, long total)
{
    long i = (long)blockIdx.x * blockDim.x + threadIdx.x;
    if (i >= total) return;
    long tok = i / NR;
    out[i] = f2b(w[tok * 2] * t1t[i] + w[tok * 2 + 1] * t1c[i]);
}

// ---------------------------------------------------------------------------
// Top-K (K=16) per token: iterated shared-memory argmax, then softmax weights
// ---------------------------------------------------------------------------
__global__ __launch_bounds__(256) void topk_kernel(
    const float* __restrict__ scores, float* __restrict__ tw,
    int* __restrict__ idx_ws, int* __restrict__ idx_out, int Nn, int Kk)
{
    int tok = blockIdx.x, tid = threadIdx.x;
    __shared__ float sc[2048];
    __shared__ float bmax[256];
    __shared__ int   bidx[256];
    __shared__ float selv[16];
    const float* sp = scores + (long)tok * Nn;
    for (int i = tid; i < Nn; i += 256) sc[i] = sp[i];
    __syncthreads();
    for (int k = 0; k < Kk; ++k) {
        float bm = -1e30f; int bi = 0;
        for (int i = tid; i < Nn; i += 256) {
            float v = sc[i];
            if (v > bm) { bm = v; bi = i; }
        }
        bmax[tid] = bm; bidx[tid] = bi;
        __syncthreads();
        for (int st = 128; st > 0; st >>= 1) {
            if (tid < st && bmax[tid + st] > bmax[tid]) {
                bmax[tid] = bmax[tid + st]; bidx[tid] = bidx[tid + st];
            }
            __syncthreads();
        }
        if (tid == 0) {
            int bi0 = bidx[0];
            selv[k] = bmax[0];
            idx_ws[(long)tok * Kk + k]  = bi0;
            idx_out[(long)tok * Kk + k] = bi0;
            sc[bi0] = -1e30f;
        }
        __syncthreads();
    }
    if (tid == 0) {
        float mx = selv[0];
        for (int k = 1; k < Kk; ++k) mx = fmaxf(mx, selv[k]);
        float e[16]; float s = 0.0f;
        for (int k = 0; k < Kk; ++k) { e[k] = __expf(selv[k] - mx); s += e[k]; }
        float inv = 1.0f / s;
        for (int k = 0; k < Kk; ++k) tw[(long)tok * Kk + k] = e[k] * inv;
    }
}

// x1 = x + sum_k tw*neurons[idx]; token_sig = bf16(sum)
__global__ __launch_bounds__(256) void gather_x1_kernel(
    const float* __restrict__ x, const float* __restrict__ neurons,
    const float* __restrict__ tw, const int* __restrict__ idx,
    float* __restrict__ x1, unsigned short* __restrict__ tsig, int Dd, int Kk)
{
    int tok = blockIdx.x;
    __shared__ int   sidx[16];
    __shared__ float stw[16];
    if (threadIdx.x < Kk) {
        sidx[threadIdx.x] = idx[(long)tok * Kk + threadIdx.x];
        stw[threadIdx.x]  = tw[(long)tok * Kk + threadIdx.x];
    }
    __syncthreads();
    for (int d = threadIdx.x; d < Dd; d += 256) {
        float acc = 0.0f;
#pragma unroll
        for (int k = 0; k < 16; ++k)
            acc += stw[k] * neurons[(long)sidx[k] * Dd + d];
        long o = (long)tok * Dd + d;
        x1[o]   = x[o] + acc;
        tsig[o] = f2b(acc);
    }
}

// sA/sB pooling per batch
__global__ __launch_bounds__(256) void sab_kernel(
    const float* __restrict__ tw, const int* __restrict__ idx,
    const float* __restrict__ coefA, const float* __restrict__ coefB,
    float* __restrict__ sA, float* __restrict__ sB, int SK, int NBc)
{
    int b = blockIdx.x, tid = threadIdx.x;
    const float* twb = tw + (long)b * SK;
    const int*   ib  = idx + (long)b * SK;
    __shared__ float red[256];
    __shared__ float acc[64];
    __shared__ float inv_s;
    float ws = 0.0f;
    for (int t = tid; t < SK; t += 256) ws += twb[t];
    red[tid] = ws; __syncthreads();
    for (int st = 128; st > 0; st >>= 1) { if (tid < st) red[tid] += red[tid + st]; __syncthreads(); }
    if (tid == 0) inv_s = 1.0f / (red[0] + 1e-8f);
    if (tid < 64) acc[tid] = 0.0f;
    __syncthreads();
    float inv = inv_s;
    float la[32], lb[32];
#pragma unroll
    for (int i = 0; i < 32; ++i) { la[i] = 0.0f; lb[i] = 0.0f; }
    for (int t = tid; t < SK; t += 256) {
        float w = twb[t] * inv;
        long id = ib[t];
        const float* ca = coefA + id * NBc;
        const float* cb = coefB + id * NBc;
#pragma unroll
        for (int i = 0; i < 32; ++i) { la[i] += w * ca[i]; lb[i] += w * cb[i]; }
    }
#pragma unroll
    for (int i = 0; i < 32; ++i) {
        atomicAdd(&acc[i], la[i]);
        atomicAdd(&acc[32 + i], lb[i]);
    }
    __syncthreads();
    if (tid < 32) {
        sA[(long)b * 32 + tid] = acc[tid];
        sB[(long)b * 32 + tid] = acc[32 + tid];
    }
}

// out[b] = bf16( sum_j s[b][j] * basis[j] )  (basis: [nb, per])
__global__ void mixbasis_kernel(const float* __restrict__ s,
    const float* __restrict__ basis, unsigned short* __restrict__ out,
    long per, int nb)
{
    long i = (long)blockIdx.x * blockDim.x + threadIdx.x;
    int b = blockIdx.y;
    if (i >= per) return;
    const float* sb = s + (long)b * nb;
    float a = 0.0f;
    for (int j = 0; j < nb; ++j) a += sb[j] * basis[(long)j * per + i];
    out[(long)b * per + i] = f2b(a);
}

__global__ void cast_kernel(const float* __restrict__ in,
    unsigned short* __restrict__ out, long n)
{
    long i = (long)blockIdx.x * blockDim.x + threadIdx.x;
    if (i < n) out[i] = f2b(in[i]);
}

// out[c*R + r] = bf16(in[r*C + c])
__global__ void transcast_kernel(const float* __restrict__ in,
    unsigned short* __restrict__ out, int R, int C)
{
    long i = (long)blockIdx.x * blockDim.x + threadIdx.x;
    if (i >= (long)R * C) return;
    int r = (int)(i / C), c = (int)(i % C);
    out[(long)c * R + r] = f2b(in[i]);
}

// ---------------------------------------------------------------------------
// Host orchestration
// ---------------------------------------------------------------------------
extern "C" void kernel_launch(void* const* d_in, const int* in_sizes, int n_in,
                              void* d_out, int out_size, void* d_ws, size_t ws_size,
                              hipStream_t stream)
{
    (void)in_sizes; (void)n_in; (void)out_size; (void)ws_size;
    const int Bc = 8, Sc = 1024, Dc = 1024, Hc = 16, Kc = 16;
    const int Nc = 2048, NRc = 32, NBc = 32, BRc = 64, FFc = 4096, RHc = 256;
    const int T = Bc * Sc;  // 8192 tokens

    const float* x      = (const float*)d_in[0];
    const float* g1     = (const float*)d_in[1];
    const float* be1    = (const float*)d_in[2];
    const float* g2     = (const float*)d_in[3];
    const float* be2    = (const float*)d_in[4];
    const float* Wq     = (const float*)d_in[5];
    const float* bq     = (const float*)d_in[6];
    const float* Wk     = (const float*)d_in[7];
    const float* bk     = (const float*)d_in[8];
    const float* Wv     = (const float*)d_in[9];
    const float* bv     = (const float*)d_in[10];
    const float* Wp     = (const float*)d_in[11];
    const float* bp     = (const float*)d_in[12];
    const float* nA     = (const float*)d_in[13];
    const float* nB     = (const float*)d_in[14];
    const float* basisA = (const float*)d_in[15];
    const float* basisB = (const float*)d_in[16];
    const float* coefA  = (const float*)d_in[17];
    const float* coefB  = (const float*)d_in[18];
    const float* Wr1    = (const float*)d_in[19];
    const float* br1    = (const float*)d_in[20];
    const float* Wr2    = (const float*)d_in[21];
    const float* br2    = (const float*)d_in[22];
    const float* Wd     = (const float*)d_in[23];
    const float* bd     = (const float*)d_in[24];

    char* wsb = (char*)d_ws;
    size_t off = 0;
    auto alloc = [&](size_t bytes) -> void* {
        off = (off + 255) & ~(size_t)255;
        void* p = wsb + off;
        off += bytes;
        return p;
    };

    unsigned short* nb16  = (unsigned short*)alloc((size_t)T * Dc * 2);
    unsigned short* Wqb   = (unsigned short*)alloc((size_t)Dc * Dc * 2);
    unsigned short* Wkb   = (unsigned short*)alloc((size_t)Dc * Dc * 2);
    unsigned short* Wvb   = (unsigned short*)alloc((size_t)Dc * Dc * 2);
    unsigned short* qb    = (unsigned short*)alloc((size_t)T * Dc * 2);
    unsigned short* kb    = (unsigned short*)alloc((size_t)T * Dc * 2);
    unsigned short* vb    = (unsigned short*)alloc((size_t)T * Dc * 2);
    unsigned short* ctxb  = (unsigned short*)alloc((size_t)T * Dc * 2);
    unsigned short* nAb   = (unsigned short*)alloc((size_t)Nc * NRc * 2);
    unsigned short* nBb   = (unsigned short*)alloc((size_t)NRc * Dc * 2);
    unsigned short* nBT   = (unsigned short*)alloc((size_t)Dc * NRc * 2);
    unsigned short* nAT   = (unsigned short*)alloc((size_t)NRc * Nc * 2);
    float*          neur  = (float*)alloc((size_t)Nc * Dc * 4);
    float*          wbuf  = (float*)alloc((size_t)T * 2 * 4);
    float*          t1t   = (float*)alloc((size_t)T * NRc * 4);
    float*          t1x   = (float*)alloc((size_t)T * NRc * 4);
    unsigned short* t1cb  = (unsigned short*)alloc((size_t)T * NRc * 2);
    float*          scrs  = (float*)alloc((size_t)T * Nc * 4);
    float*          twb   = (float*)alloc((size_t)T * Kc * 4);
    int*            idxb  = (int*)alloc((size_t)T * Kc * 4);
    float*          x1    = (float*)alloc((size_t)T * Dc * 4);
    unsigned short* tsigb = (unsigned short*)alloc((size_t)T * Dc * 2);
    unsigned short* n2b   = (unsigned short*)alloc((size_t)T * Dc * 2);
    float*          sAb   = (float*)alloc((size_t)Bc * NBc * 4);
    float*          sBb   = (float*)alloc((size_t)Bc * NBc * 4);
    unsigned short* Amb   = (unsigned short*)alloc((size_t)Bc * Dc * BRc * 2);
    unsigned short* Bmb   = (unsigned short*)alloc((size_t)Bc * BRc * FFc * 2);
    unsigned short* t2b   = (unsigned short*)alloc((size_t)Bc * Sc * BRc * 2);
    unsigned short* Wr1b  = (unsigned short*)alloc((size_t)Dc * RHc * 2);
    unsigned short* Wr2b  = (unsigned short*)alloc((size_t)RHc * FFc * 2);
    unsigned short* Wdb   = (unsigned short*)alloc((size_t)FFc * Dc * 2);
    unsigned short* hf1b  = (unsigned short*)alloc((size_t)T * RHc * 2);
    float*          Hf    = (float*)alloc((size_t)T * FFc * 4);
    unsigned short* Hb    = (unsigned short*)alloc((size_t)T * FFc * 2);

    float* outf  = (float*)d_out;
    int*   outix = (int*)(outf + (size_t)T * Dc);

    dim3 blk(256);
    auto cgrid = [](long n) { return dim3((unsigned)((n + 255) / 256)); };

    // ---- weight casts / transposed casts ----
    cast_kernel<<<cgrid((long)Dc * Dc), blk, 0, stream>>>(Wq, Wqb, (long)Dc * Dc);
    cast_kernel<<<cgrid((long)Dc * Dc), blk, 0, stream>>>(Wk, Wkb, (long)Dc * Dc);
    cast_kernel<<<cgrid((long)Dc * Dc), blk, 0, stream>>>(Wv, Wvb, (long)Dc * Dc);
    cast_kernel<<<cgrid((long)Nc * NRc), blk, 0, stream>>>(nA, nAb, (long)Nc * NRc);
    cast_kernel<<<cgrid((long)NRc * Dc), blk, 0, stream>>>(nB, nBb, (long)NRc * Dc);
    cast_kernel<<<cgrid((long)Dc * RHc), blk, 0, stream>>>(Wr1, Wr1b, (long)Dc * RHc);
    cast_kernel<<<cgrid((long)RHc * FFc), blk, 0, stream>>>(Wr2, Wr2b, (long)RHc * FFc);
    cast_kernel<<<cgrid((long)FFc * Dc), blk, 0, stream>>>(Wd, Wdb, (long)FFc * Dc);
    transcast_kernel<<<cgrid((long)NRc * Dc), blk, 0, stream>>>(nB, nBT, NRc, Dc);
    transcast_kernel<<<cgrid((long)Nc * NRc), blk, 0, stream>>>(nA, nAT, Nc, NRc);

    // ---- norm1 ----
    ln_kernel<<<dim3(T), blk, 0, stream>>>(x, g1, be1, nb16, Dc);

    // ---- QKV projections (bf16 WMMA GEMMs, 128x128 tiles) ----
    gemm_bf16_t<128,128><<<dim3(Dc / 128, T / 128, 1), blk, 0, stream>>>(
        nb16, Wqb, bq, nullptr, qb, nullptr, 0.f, 0, T, Dc, Dc, 0, 0, 0);
    gemm_bf16_t<128,128><<<dim3(Dc / 128, T / 128, 1), blk, 0, stream>>>(
        nb16, Wkb, bk, nullptr, kb, nullptr, 0.f, 0, T, Dc, Dc, 0, 0, 0);
    gemm_bf16_t<128,128><<<dim3(Dc / 128, T / 128, 1), blk, 0, stream>>>(
        nb16, Wvb, bv, nullptr, vb, nullptr, 0.f, 0, T, Dc, Dc, 0, 0, 0);

    // ---- flash attention ----
    attn_kernel<<<dim3(Sc / 64, Bc * Hc), dim3(128), 0, stream>>>(
        qb, kb, vb, ctxb, Sc, Dc, Hc);

    // ---- neurons = nA @ nB (fp32 output for gather) ----
    gemm_bf16_t<128,128><<<dim3(Dc / 128, Nc / 128, 1), blk, 0, stream>>>(
        nAb, nBb, nullptr, neur, nullptr, nullptr, 0.f, 0, Nc, Dc, NRc, 0, 0, 0);

    // ---- router gate ----
    wp_kernel<<<dim3(T), blk, 0, stream>>>(nb16, ctxb, Wp, bp, wbuf, Dc);

    // ---- low-rank scores: t1 = n@nB^T, ctx@nB^T; combine; @ nA^T ----
    gemm_bf16_t<64,64><<<dim3(1, T / 64, 1), blk, 0, stream>>>(
        nb16, nBT, nullptr, t1t, nullptr, nullptr, 0.f, 0, T, NRc, Dc, 0, 0, 0);
    gemm_bf16_t<64,64><<<dim3(1, T / 64, 1), blk, 0, stream>>>(
        ctxb, nBT, nullptr, t1x, nullptr, nullptr, 0.f, 0, T, NRc, Dc, 0, 0, 0);
    combine_t1_kernel<<<cgrid((long)T * NRc), blk, 0, stream>>>(
        t1t, t1x, wbuf, t1cb, NRc, (long)T * NRc);
    gemm_bf16_t<128,128><<<dim3(Nc / 128, T / 128, 1), blk, 0, stream>>>(
        t1cb, nAT, nullptr, scrs, nullptr, nullptr, 0.f, 0, T, Nc, NRc, 0, 0, 0);

    // ---- top-k + routing weights (idx also to d_out tail) ----
    topk_kernel<<<dim3(T), blk, 0, stream>>>(scrs, twb, idxb, outix, Nc, Kc);

    // ---- x1 = x + routed neurons; token_sig ----
    gather_x1_kernel<<<dim3(T), blk, 0, stream>>>(x, neur, twb, idxb, x1, tsigb, Dc, Kc);

    // ---- norm2 ----
    ln_kernel<<<dim3(T), blk, 0, stream>>>(x1, g2, be2, n2b, Dc);

    // ---- basis coefficients + per-batch low-rank factors ----
    sab_kernel<<<dim3(Bc), blk, 0, stream>>>(twb, idxb, coefA, coefB, sAb, sBb, Sc * Kc, NBc);
    mixbasis_kernel<<<dim3((unsigned)(((long)Dc * BRc + 255) / 256), Bc), blk, 0, stream>>>(
        sAb, basisA, Amb, (long)Dc * BRc, NBc);
    mixbasis_kernel<<<dim3((unsigned)(((long)BRc * FFc + 255) / 256), Bc), blk, 0, stream>>>(
        sBb, basisB, Bmb, (long)BRc * FFc, NBc);

    // ---- t2 = n2 @ A  (batched, N=BR=64) ----
    gemm_bf16_t<64,64><<<dim3(1, Sc / 64, Bc), blk, 0, stream>>>(
        n2b, Amb, nullptr, nullptr, t2b, nullptr, 0.f, 0,
        Sc, BRc, Dc, (long)Sc * Dc, (long)Dc * BRc, (long)Sc * BRc);

    // ---- h_fine = relu(token_sig @ Wr1 + br1) @ Wr2 + br2 ----
    gemm_bf16_t<128,128><<<dim3(RHc / 128, T / 128, 1), blk, 0, stream>>>(
        tsigb, Wr1b, br1, nullptr, hf1b, nullptr, 0.f, 2, T, RHc, Dc, 0, 0, 0);
    gemm_bf16_t<128,128><<<dim3(FFc / 128, T / 128, 1), blk, 0, stream>>>(
        hf1b, Wr2b, br2, Hf, nullptr, nullptr, 0.f, 0, T, FFc, RHc, 0, 0, 0);

    // ---- h = bf16( gelu(t2 @ Bm) + 0.1 * h_fine )  (batched) ----
    gemm_bf16_t<128,128><<<dim3(FFc / 128, Sc / 128, Bc), blk, 0, stream>>>(
        t2b, Bmb, nullptr, nullptr, Hb, Hf, 0.1f, 1,
        Sc, FFc, BRc, (long)Sc * BRc, (long)BRc * FFc, (long)Sc * FFc);

    // ---- out = x1 + h @ Wd + bd  (fp32 into d_out) ----
    gemm_bf16_t<128,128><<<dim3(Dc / 128, T / 128, 1), blk, 0, stream>>>(
        Hb, Wdb, bd, outf, nullptr, x1, 1.0f, 0, T, Dc, FFc, 0, 0, 0);
}
